// OrbitalFeatureGNN_42374147342432
// MI455X (gfx1250) — matrix operand order repaired and checked by reading, not compiled, not run on hardware
//
#include <hip/hip_runtime.h>
#include <hip/hip_bf16.h>
#include <math.h>

typedef __attribute__((ext_vector_type(16))) _Float16 v16h;
typedef __attribute__((ext_vector_type(8)))  _Float16 v8h;
typedef __attribute__((ext_vector_type(8)))  float    v8f;

#define BB   16
#define SS   2
#define NN   16
#define OO   32
#define FF   64
#define EE   128
#define HH   128
#define NIT  2

// ---- WMMA helpers (wave32, 16x16x32 f16 -> f32 acc) ----

__device__ __forceinline__ v8f wmma_f16(v16h a, v16h b, v8f c) {
    return __builtin_amdgcn_wmma_f32_16x16x32_f16(
        /*neg_a=*/false, a, /*neg_b=*/false, b,
        /*c_mod=*/(short)0, c, /*reuse_a=*/false, /*reuse_b=*/false);
}

__device__ __forceinline__ v16h cat8(v8h lo, v8h hi) {
    return __builtin_shufflevector(lo, hi, 0, 1, 2, 3, 4, 5, 6, 7,
                                           8, 9, 10, 11, 12, 13, 14, 15);
}

// A fragment (16x32 tile) from row-major f16 with leading dim ld.
// Lane (h=lane>>4, m=lane&15): halves = row m, k = {k0+8h..+7, k0+16+8h..+7}
// -> two contiguous 16-byte loads per lane.
__device__ __forceinline__ v16h load_a_h(const _Float16* src, int ld, int m0, int k0, int lane) {
    int h = lane >> 4, m = lane & 15;
    const _Float16* row = src + (size_t)(m0 + m) * ld + k0 + 8 * h;
    v8h lo = *(const v8h*)row;
    v8h hi = *(const v8h*)(row + 16);
    return cat8(lo, hi);
}

// B fragment from fragment-linear pre-swizzled f16 weights:
// dst[((kt*NT + nt)*32 + lane)*16 + e] == W[kt*32 + 8*(lane>>4) + (e&7) + 16*(e>>3)][nt*16 + (lane&15)]
__device__ __forceinline__ v16h load_b_sw(const _Float16* wsw, int NT, int kt, int nt, int lane) {
    const _Float16* p = wsw + ((((size_t)kt * NT + nt) * 32 + lane) << 4);
    v8h lo = *(const v8h*)p;
    v8h hi = *(const v8h*)(p + 8);
    return cat8(lo, hi);
}

// C/D: 16x16 f32; VGPR r holds row (r + 8*laneHalf), col = lane&15.
__device__ __forceinline__ void store_c(float* dst, int ld, int m0, int n0, int lane, v8f acc) {
    int n = lane & 15, half = lane >> 4;
#pragma unroll
    for (int r = 0; r < 8; ++r)
        dst[(size_t)(m0 + r + 8 * half) * ld + n0 + n] = acc[r];
}

// ---- Kernel 0: convert+swizzle a weight matrix (Ktot x Ntot f32, row-major)
// into WMMA-B fragment-linear f16 layout. ----
__global__ __launch_bounds__(256) void swizzle_w_kernel(
    const float* __restrict__ src, _Float16* __restrict__ dst, int Ktot, int Ntot)
{
    int NT = Ntot >> 4;
    size_t idx = (size_t)blockIdx.x * 256 + threadIdx.x;   // == Ktot*Ntot total
    int e = (int)(idx & 15);
    int l = (int)((idx >> 4) & 31);
    size_t rest = idx >> 9;
    int nt = (int)(rest % NT);
    int kt = (int)(rest / NT);
    int k = kt * 32 + 8 * (l >> 4) + (e & 7) + 16 * (e >> 3);
    int n = nt * 16 + (l & 15);
    dst[idx] = (_Float16)src[(size_t)k * Ntot + n];
}

// ---- Kernel 1: edge features (rbf * [1, dx, dy, dz]) -> f16 (GEMM A operand) ----
__global__ __launch_bounds__(128) void edges_kernel(
    const float* __restrict__ dist, const float* __restrict__ diff,
    _Float16* __restrict__ edgesH)
{
    int p = blockIdx.x;           // b*256 + i*16 + j  (B*N*N = 4096)
    int e = threadIdx.x;          // 0..127
    float r = dist[p];
    int c = e >> 5, k = e & 31;
    float q = (float)k * (1.0f / 31.0f);
    float mu = q * q * 5.0f;
    float sigma = (1.0f + 5.0f * q) * (1.0f / 7.0f);
    float t = (r - mu) / sigma;
    float val = r * r * expf(-r - t * t);
    float fac = (c == 0) ? 1.0f : diff[(size_t)p * 3 + (c - 1)];
    edgesH[(size_t)p * EE + e] = (_Float16)(val * fac);
}

// ---- Kernel 2: Eterm[it,b] = edges[b] (256x128) @ Wm1[it][128:256,:] + bm1 ----
__global__ __launch_bounds__(256) void eterm_kernel(
    const _Float16* __restrict__ edgesH, const _Float16* __restrict__ Wm1sw,
    const float* __restrict__ bm1, float* __restrict__ eterm)
{
    int blk = blockIdx.x;               // it*B + b
    int it = blk / BB, b = blk % BB;
    const _Float16* eb = edgesH + (size_t)b * 256 * EE;
    const _Float16* w  = Wm1sw + (size_t)it * 256 * HH;   // swizzled; kt 4..7 = rows 128..255
    float* out = eterm + (size_t)blk * 256 * HH;
    int wave = threadIdx.x >> 5, lane = threadIdx.x & 31;
    for (int t = wave; t < 128; t += 8) {                  // 16 m-tiles x 8 n-tiles
        int mt = t >> 3, nt = t & 7;
        v8f acc = {};
#pragma unroll
        for (int kt = 0; kt < 4; ++kt) {
            v16h a  = load_a_h(eb, EE, mt * 16, kt * 32, lane);
            v16h bf = load_b_sw(w, 8, 4 + kt, nt, lane);
            acc = wmma_f16(a, bf, acc);
        }
        int n = lane & 15, half = lane >> 4;
        float bias = bm1[it * HH + nt * 16 + n];
#pragma unroll
        for (int r = 0; r < 8; ++r)
            out[(size_t)(mt * 16 + r + 8 * half) * HH + nt * 16 + n] = acc[r] + bias;
    }
}

// ---- Kernel 3: full 2-iteration GNN for one (b,s,o) slice ----
__global__ __launch_bounds__(128) void gnn_kernel(
    const float* __restrict__ orb, const float* __restrict__ eterm,
    const _Float16* __restrict__ Wm1sw, const _Float16* __restrict__ Wm2sw,
    const _Float16* __restrict__ Wu1sw, const _Float16* __restrict__ Wu2sw,
    const float* __restrict__ bm2, const float* __restrict__ bu1,
    const float* __restrict__ bu2,
    float* __restrict__ y, int flip)
{
    __shared__ float nodes[NN][FF];                       // f32 master (residual)
    __shared__ __align__(16) _Float16 nodesH[NN][FF];     // A operand
    __shared__ float A1[NN][HH];
    __shared__ float A2[NN][HH];
    __shared__ __align__(16) _Float16 TmH[NN][HH];        // A operand
    __shared__ float agg[NN][FF];
    __shared__ __align__(16) _Float16 updInH[NN][HH];     // A operand
    __shared__ __align__(16) _Float16 h2H[NN][HH];        // A operand

    int slice = blockIdx.x;             // b*S*O + s*O + o
    int b = slice / (SS * OO);
    int s = (slice / OO) % SS;
    int o = slice % OO;
    int ssrc = flip ? (SS - 1 - s) : s;
    int tid = threadIdx.x;
    int wave = tid >> 5, lane = tid & 31;

    // nodes0[b,s,o,n,f] = orb[b, ssrc, n, o, f]
    for (int idx = tid; idx < NN * FF; idx += 128) {
        int n = idx >> 6, f = idx & 63;
        nodes[n][f] = orb[((((size_t)b * SS + ssrc) * NN + n) * OO + o) * FF + f];
    }
    __syncthreads();

    for (int it = 0; it < NIT; ++it) {
        const _Float16* wm1 = Wm1sw + (size_t)it * 256 * HH;
        const float* ebase = eterm + (size_t)(it * BB + b) * 256 * HH;
        if (tid < 16)
            __builtin_prefetch(ebase + (size_t)tid * 16 * HH, 0, 0);

        // refresh f16 mirror of nodes
        for (int idx = tid; idx < NN * FF; idx += 128) {
            int n = idx >> 6, f = idx & 63;
            nodesH[n][f] = (_Float16)nodes[n][f];
        }
        __syncthreads();

        // A1 = nodes @ Wm1[0:64,:], A2 = nodes @ Wm1[64:128,:]   (each 16x128)
        for (int t = wave; t < 16; t += 4) {
            int mat = t >> 3;          // 0 -> A1 (kt 0..1), 1 -> A2 (kt 2..3)
            int nt  = t & 7;
            v8f acc = {};
#pragma unroll
            for (int kt = 0; kt < 2; ++kt) {
                v16h a  = load_a_h(&nodesH[0][0], FF, 0, kt * 32, lane);
                v16h bf = load_b_sw(wm1, 8, mat * 2 + kt, nt, lane);
                acc = wmma_f16(a, bf, acc);
            }
            store_c(mat ? &A2[0][0] : &A1[0][0], HH, 0, nt * 16, lane, acc);
        }
        __syncthreads();

        // Tm[i][h] = (1/16) * sum_j tanh(A1[i][h] + A2[j][h] + Eterm[i,j,h])  (bm1 folded)
        for (int idx = tid; idx < NN * HH; idx += 128) {
            int i = idx >> 7, h = idx & 127;
            float base = A1[i][h];
            const float* ep = ebase + ((size_t)i * 16) * HH + h;
            float sum = 0.0f;
#pragma unroll
            for (int j = 0; j < 16; ++j)
                sum += tanhf(base + A2[j][h] + ep[(size_t)j * HH]);
            TmH[i][h] = (_Float16)(sum * (1.0f / 16.0f));
        }
        __syncthreads();

        // agg = Tm @ Wm2 + bm2   (16x128 @ 128x64)
        {
            int nt = wave;             // 0..3
            v8f acc = {};
#pragma unroll
            for (int kt = 0; kt < 4; ++kt) {
                v16h a  = load_a_h(&TmH[0][0], HH, 0, kt * 32, lane);
                v16h bf = load_b_sw(Wm2sw + (size_t)it * HH * FF, 4, kt, nt, lane);
                acc = wmma_f16(a, bf, acc);
            }
            int n = lane & 15, half = lane >> 4;
            float bias = bm2[it * FF + nt * 16 + n];
#pragma unroll
            for (int r = 0; r < 8; ++r)
                agg[r + 8 * half][nt * 16 + n] = acc[r] + bias;
        }
        __syncthreads();

        // updIn = [nodes | agg]  -> f16 A operand
        for (int idx = tid; idx < NN * HH; idx += 128) {
            int i = idx >> 7, c = idx & 127;
            updInH[i][c] = (_Float16)((c < FF) ? nodes[i][c] : agg[i][c - FF]);
        }
        __syncthreads();

        // h2 = tanh(updIn @ Wu1 + bu1)   (16x128 @ 128x128)
        for (int nt = wave; nt < 8; nt += 4) {
            v8f acc = {};
#pragma unroll
            for (int kt = 0; kt < 4; ++kt) {
                v16h a  = load_a_h(&updInH[0][0], HH, 0, kt * 32, lane);
                v16h bf = load_b_sw(Wu1sw + (size_t)it * HH * HH, 8, kt, nt, lane);
                acc = wmma_f16(a, bf, acc);
            }
            int n = lane & 15, half = lane >> 4;
            float bias = bu1[it * HH + nt * 16 + n];
#pragma unroll
            for (int r = 0; r < 8; ++r)
                h2H[r + 8 * half][nt * 16 + n] = (_Float16)tanhf(acc[r] + bias);
        }
        __syncthreads();

        // nodes += h2 @ Wu2 + bu2   (16x128 @ 128x64)
        {
            int nt = wave;             // 0..3
            v8f acc = {};
#pragma unroll
            for (int kt = 0; kt < 4; ++kt) {
                v16h a  = load_a_h(&h2H[0][0], HH, 0, kt * 32, lane);
                v16h bf = load_b_sw(Wu2sw + (size_t)it * HH * FF, 4, kt, nt, lane);
                acc = wmma_f16(a, bf, acc);
            }
            int n = lane & 15, half = lane >> 4;
            float bias = bu2[it * FF + nt * 16 + n];
#pragma unroll
            for (int r = 0; r < 8; ++r) {
                int i = r + 8 * half;
                nodes[i][nt * 16 + n] += acc[r] + bias;
            }
        }
        __syncthreads();
    }

    // y laid out (B,S,O,N,F) flat
    for (int idx = tid; idx < NN * FF; idx += 128) {
        int n = idx >> 6, f = idx & 63;
        y[((size_t)slice * NN + n) * FF + f] = nodes[n][f];
    }
}

// ---- Kernel 4: antisymmetrize + moveaxis back to (B,S,N,O,F) ----
__global__ __launch_bounds__(256) void combine_kernel(
    const float* __restrict__ yA, const float* __restrict__ yB,
    float* __restrict__ out)
{
    size_t idx = (size_t)blockIdx.x * 256 + threadIdx.x;   // over B*S*N*O*F
    int f = idx & 63; size_t r = idx >> 6;
    int o = r & 31;   r >>= 5;
    int n = r & 15;   r >>= 4;
    int s = (int)(r & 1); r >>= 1;
    int b = (int)r;
    size_t ia = ((((size_t)b * SS + s)       * OO + o) * NN + n) * FF + f;
    size_t ib = ((((size_t)b * SS + (1 - s)) * OO + o) * NN + n) * FF + f;
    out[idx] = 0.5f * (yA[ia] - yB[ib]);
}

extern "C" void kernel_launch(void* const* d_in, const int* in_sizes, int n_in,
                              void* d_out, int out_size, void* d_ws, size_t ws_size,
                              hipStream_t stream) {
    const float* orb  = (const float*)d_in[0];
    const float* diff = (const float*)d_in[1];
    const float* dist = (const float*)d_in[2];
    const float* Wm1  = (const float*)d_in[3];
    const float* bm1  = (const float*)d_in[4];
    const float* Wm2  = (const float*)d_in[5];
    const float* bm2  = (const float*)d_in[6];
    const float* Wu1  = (const float*)d_in[7];
    const float* bu1  = (const float*)d_in[8];
    const float* Wu2  = (const float*)d_in[9];
    const float* bu2  = (const float*)d_in[10];
    float* out = (float*)d_out;

    char* ws = (char*)d_ws;
    _Float16* edgesH = (_Float16*)(ws);                  // 1 MB  (B*256*128 f16)
    float* eterm = (float*)(ws + ((size_t)1 << 20));     // 4 MB  (NIT*B*256*128 f32)
    float* yA    = (float*)(ws + ((size_t)5 << 20));     // 4 MB
    float* yB    = (float*)(ws + ((size_t)9 << 20));     // 4 MB
    _Float16* wsw = (_Float16*)(ws + ((size_t)13 << 20)); // 256 KB f16 swizzled weights
    _Float16* Wm1sw = wsw;                               // 2 * 256*128
    _Float16* Wm2sw = Wm1sw + 2 * 256 * HH;              // 2 * 128*64
    _Float16* Wu1sw = Wm2sw + 2 * HH * FF;               // 2 * 128*128
    _Float16* Wu2sw = Wu1sw + 2 * HH * HH;               // 2 * 128*64

    for (int it = 0; it < NIT; ++it) {
        swizzle_w_kernel<<<(256 * HH) / 256, 256, 0, stream>>>(
            Wm1 + (size_t)it * 256 * HH, Wm1sw + (size_t)it * 256 * HH, 256, HH);
        swizzle_w_kernel<<<(HH * FF) / 256, 256, 0, stream>>>(
            Wm2 + (size_t)it * HH * FF, Wm2sw + (size_t)it * HH * FF, HH, FF);
        swizzle_w_kernel<<<(HH * HH) / 256, 256, 0, stream>>>(
            Wu1 + (size_t)it * HH * HH, Wu1sw + (size_t)it * HH * HH, HH, HH);
        swizzle_w_kernel<<<(HH * FF) / 256, 256, 0, stream>>>(
            Wu2 + (size_t)it * HH * FF, Wu2sw + (size_t)it * HH * FF, HH, FF);
    }

    edges_kernel<<<BB * NN * NN, 128, 0, stream>>>(dist, diff, edgesH);
    eterm_kernel<<<NIT * BB, 256, 0, stream>>>(edgesH, Wm1sw, bm1, eterm);
    gnn_kernel<<<BB * SS * OO, 128, 0, stream>>>(orb, eterm, Wm1sw, Wm2sw, Wu1sw, Wu2sw,
                                                 bm2, bu1, bu2, yA, 0);
    gnn_kernel<<<BB * SS * OO, 128, 0, stream>>>(orb, eterm, Wm1sw, Wm2sw, Wu1sw, Wu2sw,
                                                 bm2, bu1, bu2, yB, 1);
    combine_kernel<<<(BB * SS * NN * OO * FF) / 256, 256, 0, stream>>>(yA, yB, out);
}
